// BertAttention_88742614269968
// MI455X (gfx1250) — matrix operand order repaired and checked by reading, not compiled, not run on hardware
//
#include <hip/hip_runtime.h>
#include <hip/hip_bf16.h>

#define HID 1024
#define SEQ 2048
#define NH  16
#define HD  64

typedef __attribute__((ext_vector_type(16))) _Float16 v16h;
typedef __attribute__((ext_vector_type(8)))  float    v8f;
typedef __attribute__((ext_vector_type(4)))  float    v4f;

union FragU { v16h v; v4f q[2]; };

// ---- A fragment: 16x32 f16 tile, row-major source, leading dim ld ----
// lane L: row = L%16; K-halves: lanes 0-15 hold K[0..7]+K[16..23],
// lanes 16-31 hold K[8..15]+K[24..31]  (ISA 7.12.2, 16-bit A 16x32)
__device__ __forceinline__ v16h load_frag_a(const _Float16* base, int ld) {
    const int lane = threadIdx.x & 31;
    const int m  = lane & 15;
    const int kb = (lane >> 4) << 3;          // 0 or 8
    const _Float16* p = base + m * ld + kb;
    FragU u;
    u.q[0] = *(const v4f*)(p);                // K = kb .. kb+7
    u.q[1] = *(const v4f*)(p + 16);           // K = kb+16 .. kb+23
    return u.v;
}

// ---- B fragment: 32x16 f16 tile whose storage is B^T row-major ----
// base = &Bt[n0][k0], leading dim ld. lane L: col = L%16;
// lanes 0-15 hold K[0..7]+K[16..23], lanes 16-31 hold K[8..15]+K[24..31].
__device__ __forceinline__ v16h load_frag_b(const _Float16* base, int ld) {
    const int lane = threadIdx.x & 31;
    const int n  = lane & 15;
    const int kb = (lane >> 4) << 3;          // 0 or 8
    const _Float16* p = base + n * ld + kb;
    FragU u;
    u.q[0] = *(const v4f*)(p);
    u.q[1] = *(const v4f*)(p + 16);
    return u.v;
}

// ======================= conversion kernels =======================
__global__ void cvt_f32_to_f16(const float* __restrict__ src,
                               _Float16* __restrict__ dst, int n) {
    int i = blockIdx.x * blockDim.x + threadIdx.x;
    if (i < n) dst[i] = (_Float16)src[i];
}

// ======================= fused QKV WMMA GEMM =======================
// grid: (N/256, M/128, 3); 256 threads = 8 waves; wave tile 64x64
__global__ void __launch_bounds__(256, 1)
gemm_qkv_kernel(const _Float16* __restrict__ X,
                const _Float16* __restrict__ Wq16,
                const _Float16* __restrict__ Wk16,
                const _Float16* __restrict__ Wv16,
                const float* __restrict__ bq,
                const float* __restrict__ bk,
                const float* __restrict__ bv,
                _Float16* __restrict__ Qo,    // [B,NH,S,HD], pre-scaled
                _Float16* __restrict__ Ko,    // [B,NH,S,HD]
                _Float16* __restrict__ Vt)    // [B,NH,HD,S] (transposed)
{
    const int mode = blockIdx.z;
    const _Float16* W = (mode == 0) ? Wq16 : (mode == 1) ? Wk16 : Wv16;
    const float* bias = (mode == 0) ? bq : (mode == 1) ? bk : bv;

    const int wid  = threadIdx.x >> 5;
    const int lane = threadIdx.x & 31;
    const int m0 = blockIdx.y * 128 + (wid >> 2) * 64;
    const int n0 = blockIdx.x * 256 + (wid & 3) * 64;

    v8f zero = {};
    v8f acc[4][4];
    #pragma unroll
    for (int i = 0; i < 4; ++i)
        #pragma unroll
        for (int j = 0; j < 4; ++j) acc[i][j] = zero;

    for (int k = 0; k < HID; k += 32) {
        v16h af[4], bf[4];
        #pragma unroll
        for (int mt = 0; mt < 4; ++mt)
            af[mt] = load_frag_a(X + (size_t)(m0 + mt * 16) * HID + k, HID);
        #pragma unroll
        for (int nt = 0; nt < 4; ++nt)
            bf[nt] = load_frag_b(W + (size_t)(n0 + nt * 16) * HID + k, HID);
        #pragma unroll
        for (int mt = 0; mt < 4; ++mt)
            #pragma unroll
            for (int nt = 0; nt < 4; ++nt)
                acc[mt][nt] = __builtin_amdgcn_wmma_f32_16x16x32_f16(
                    false, af[mt], false, bf[nt], (short)0, acc[mt][nt],
                    false, false);
    }

    const int col   = lane & 15;
    const int rplus = (lane >> 4) << 3;       // 0 or 8
    #pragma unroll
    for (int mt = 0; mt < 4; ++mt)
        #pragma unroll
        for (int nt = 0; nt < 4; ++nt)
            #pragma unroll
            for (int r = 0; r < 8; ++r) {
                int m = m0 + mt * 16 + r + rplus;
                int n = n0 + nt * 16 + col;
                float v = acc[mt][nt][r] + bias[n];
                int b = m >> 11, s = m & (SEQ - 1);
                int h = n >> 6,  d = n & (HD - 1);
                if (mode == 0)
                    Qo[(((size_t)(b * NH + h) * SEQ) + s) * HD + d] =
                        (_Float16)(v * 0.125f);        // fold 1/sqrt(64)
                else if (mode == 1)
                    Ko[(((size_t)(b * NH + h) * SEQ) + s) * HD + d] = (_Float16)v;
                else
                    Vt[(((size_t)(b * NH + h) * HD) + d) * SEQ + s] = (_Float16)v;
            }
}

// ======================= flash attention (WMMA) =======================
// grid: (S/64, NH, B); 32 threads (1 wave). 64 q-rows per wave.
__global__ void __launch_bounds__(32, 1)
attn_kernel(const _Float16* __restrict__ Q,
            const _Float16* __restrict__ K,
            const _Float16* __restrict__ Vt,
            const float* __restrict__ mask,     // [B,1,1,S]
            _Float16* __restrict__ CTX)         // [B,S,HID] merged heads
{
    __shared__ __align__(16) _Float16 pbuf[64 * 32];   // P tile, 4 KB

    const int lane = threadIdx.x & 31;
    const int qt = blockIdx.x, h = blockIdx.y, b = blockIdx.z;
    const int bh = b * NH + h;
    const _Float16* Qh = Q  + (size_t)bh * SEQ * HD + (size_t)qt * 64 * HD;
    const _Float16* Kh = K  + (size_t)bh * SEQ * HD;
    const _Float16* Vh = Vt + (size_t)bh * HD * SEQ;
    const float*  mrow = mask + (size_t)b * SEQ;

    // resident Q fragments: 4 M-tiles x 2 K-steps (d = 64)
    v16h qf[4][2];
    #pragma unroll
    for (int mt = 0; mt < 4; ++mt)
        #pragma unroll
        for (int ks = 0; ks < 2; ++ks)
            qf[mt][ks] = load_frag_a(Qh + mt * 16 * HD + ks * 32, HD);

    v8f zero = {};
    v8f cacc[4][4];
    #pragma unroll
    for (int i = 0; i < 4; ++i)
        #pragma unroll
        for (int j = 0; j < 4; ++j) cacc[i][j] = zero;

    float mst[4][8], lst[4][8];
    #pragma unroll
    for (int i = 0; i < 4; ++i)
        #pragma unroll
        for (int r = 0; r < 8; ++r) { mst[i][r] = -1e30f; lst[i][r] = 0.f; }

    const int rplus = (lane >> 4) << 3;       // 0 or 8
    const int col   = lane & 15;

    for (int kb = 0; kb < SEQ / 32; ++kb) {
        // ---- scores: S = Q x K^T  (K stored [s,d] == B^T row-major) ----
        v16h kf[2][2];
        #pragma unroll
        for (int nt = 0; nt < 2; ++nt)
            #pragma unroll
            for (int ks = 0; ks < 2; ++ks)
                kf[nt][ks] = load_frag_b(Kh + (size_t)(kb * 32 + nt * 16) * HD + ks * 32, HD);

        v8f sc[4][2];
        #pragma unroll
        for (int mt = 0; mt < 4; ++mt)
            #pragma unroll
            for (int nt = 0; nt < 2; ++nt) sc[mt][nt] = zero;
        #pragma unroll
        for (int ks = 0; ks < 2; ++ks)
            #pragma unroll
            for (int mt = 0; mt < 4; ++mt)
                #pragma unroll
                for (int nt = 0; nt < 2; ++nt)
                    sc[mt][nt] = __builtin_amdgcn_wmma_f32_16x16x32_f16(
                        false, qf[mt][ks], false, kf[nt][ks], (short)0,
                        sc[mt][nt], false, false);

        float mv0 = mrow[kb * 32 + col];
        float mv1 = mrow[kb * 32 + 16 + col];

        // ---- online softmax (row spans 16 lanes of a half-wave) ----
        #pragma unroll
        for (int mt = 0; mt < 4; ++mt)
            #pragma unroll
            for (int r = 0; r < 8; ++r) {
                float s0 = sc[mt][0][r] + mv0;
                float s1 = sc[mt][1][r] + mv1;
                float mx = fmaxf(s0, s1);
                #pragma unroll
                for (int off = 1; off < 16; off <<= 1)
                    mx = fmaxf(mx, __shfl_xor(mx, off, 32));
                float mnew = fmaxf(mst[mt][r], mx);
                float corr = __expf(mst[mt][r] - mnew);
                float p0 = __expf(s0 - mnew);
                float p1 = __expf(s1 - mnew);
                float ps = p0 + p1;
                #pragma unroll
                for (int off = 1; off < 16; off <<= 1)
                    ps += __shfl_xor(ps, off, 32);
                lst[mt][r] = lst[mt][r] * corr + ps;
                mst[mt][r] = mnew;
                #pragma unroll
                for (int nt = 0; nt < 4; ++nt) cacc[mt][nt][r] *= corr;
                int row = mt * 16 + r + rplus;
                pbuf[row * 32 + col]      = (_Float16)p0;
                pbuf[row * 32 + 16 + col] = (_Float16)p1;
            }
        __syncthreads();

        // ---- ctx += P x V  (P from LDS as A-frag; V^T rows contiguous) ----
        v16h pf[4], vf[4];
        #pragma unroll
        for (int mt = 0; mt < 4; ++mt)
            pf[mt] = load_frag_a(pbuf + mt * 16 * 32, 32);
        #pragma unroll
        for (int dt = 0; dt < 4; ++dt)
            vf[dt] = load_frag_b(Vh + (size_t)(dt * 16) * SEQ + kb * 32, SEQ);
        #pragma unroll
        for (int mt = 0; mt < 4; ++mt)
            #pragma unroll
            for (int dt = 0; dt < 4; ++dt)
                cacc[mt][dt] = __builtin_amdgcn_wmma_f32_16x16x32_f16(
                    false, pf[mt], false, vf[dt], (short)0, cacc[mt][dt],
                    false, false);
        __syncthreads();
    }

    // ---- epilogue: normalize and store merged-head context (f16) ----
    #pragma unroll
    for (int mt = 0; mt < 4; ++mt)
        #pragma unroll
        for (int r = 0; r < 8; ++r) {
            float inv = 1.0f / lst[mt][r];
            int srow = qt * 64 + mt * 16 + r + rplus;
            #pragma unroll
            for (int dt = 0; dt < 4; ++dt) {
                int d = dt * 16 + col;
                CTX[((size_t)(b * SEQ + srow)) * HID + h * HD + d] =
                    (_Float16)(cacc[mt][dt][r] * inv);
            }
        }
}

// ============ output projection GEMM + bias + residual (f32) ============
__global__ void __launch_bounds__(256, 1)
gemm_out_kernel(const _Float16* __restrict__ CTX,
                const _Float16* __restrict__ Wo16,
                const float* __restrict__ bo,
                const float* __restrict__ hidden,
                float* __restrict__ outpre)
{
    const int wid  = threadIdx.x >> 5;
    const int lane = threadIdx.x & 31;
    const int m0 = blockIdx.y * 128 + (wid >> 2) * 64;
    const int n0 = blockIdx.x * 256 + (wid & 3) * 64;

    v8f zero = {};
    v8f acc[4][4];
    #pragma unroll
    for (int i = 0; i < 4; ++i)
        #pragma unroll
        for (int j = 0; j < 4; ++j) acc[i][j] = zero;

    for (int k = 0; k < HID; k += 32) {
        v16h af[4], bf[4];
        #pragma unroll
        for (int mt = 0; mt < 4; ++mt)
            af[mt] = load_frag_a(CTX + (size_t)(m0 + mt * 16) * HID + k, HID);
        #pragma unroll
        for (int nt = 0; nt < 4; ++nt)
            bf[nt] = load_frag_b(Wo16 + (size_t)(n0 + nt * 16) * HID + k, HID);
        #pragma unroll
        for (int mt = 0; mt < 4; ++mt)
            #pragma unroll
            for (int nt = 0; nt < 4; ++nt)
                acc[mt][nt] = __builtin_amdgcn_wmma_f32_16x16x32_f16(
                    false, af[mt], false, bf[nt], (short)0, acc[mt][nt],
                    false, false);
    }

    const int col   = lane & 15;
    const int rplus = (lane >> 4) << 3;
    #pragma unroll
    for (int mt = 0; mt < 4; ++mt)
        #pragma unroll
        for (int nt = 0; nt < 4; ++nt)
            #pragma unroll
            for (int r = 0; r < 8; ++r) {
                int m = m0 + mt * 16 + r + rplus;
                int n = n0 + nt * 16 + col;
                size_t idx = (size_t)m * HID + n;
                outpre[idx] = acc[mt][nt][r] + bo[n] + hidden[idx];
            }
}

// ======================= LayerNorm =======================
__global__ void __launch_bounds__(256, 1)
ln_kernel(const float* __restrict__ X,
          const float* __restrict__ gamma,
          const float* __restrict__ beta,
          float* __restrict__ out)
{
    __shared__ float red[256];
    const int row = blockIdx.x;
    const int tid = threadIdx.x;
    const float* x = X + (size_t)row * HID;

    float v[4]; float s = 0.f;
    #pragma unroll
    for (int i = 0; i < 4; ++i) { v[i] = x[tid * 4 + i]; s += v[i]; }
    red[tid] = s; __syncthreads();
    for (int off = 128; off > 0; off >>= 1) {
        if (tid < off) red[tid] += red[tid + off];
        __syncthreads();
    }
    float mu = red[0] * (1.0f / HID);
    __syncthreads();

    float vs = 0.f;
    #pragma unroll
    for (int i = 0; i < 4; ++i) { float d = v[i] - mu; vs += d * d; }
    red[tid] = vs; __syncthreads();
    for (int off = 128; off > 0; off >>= 1) {
        if (tid < off) red[tid] += red[tid + off];
        __syncthreads();
    }
    float rstd = rsqrtf(red[0] * (1.0f / HID) + 1e-12f);

    #pragma unroll
    for (int i = 0; i < 4; ++i) {
        int c = tid * 4 + i;
        out[(size_t)row * HID + c] = (v[i] - mu) * rstd * gamma[c] + beta[c];
    }
}

// ======================= host launcher =======================
extern "C" void kernel_launch(void* const* d_in, const int* in_sizes, int n_in,
                              void* d_out, int out_size, void* d_ws, size_t ws_size,
                              hipStream_t stream) {
    const float* hidden = (const float*)d_in[0];
    const float* mask   = (const float*)d_in[1];
    const float* Wq = (const float*)d_in[2];
    const float* bq = (const float*)d_in[3];
    const float* Wk = (const float*)d_in[4];
    const float* bk = (const float*)d_in[5];
    const float* Wv = (const float*)d_in[6];
    const float* bv = (const float*)d_in[7];
    const float* Wo = (const float*)d_in[8];
    const float* bo = (const float*)d_in[9];
    const float* ln_g = (const float*)d_in[10];
    const float* ln_b = (const float*)d_in[11];

    const size_t MTOK = 4ull * SEQ;           // 8192 tokens
    size_t off = 0;
    auto alloc = [&](size_t bytes) -> void* {
        void* p = (char*)d_ws + off;
        off += (bytes + 255) & ~(size_t)255;
        return p;
    };
    _Float16* X16  = (_Float16*)alloc(MTOK * HID * 2);
    _Float16* Wq16 = (_Float16*)alloc((size_t)HID * HID * 2);
    _Float16* Wk16 = (_Float16*)alloc((size_t)HID * HID * 2);
    _Float16* Wv16 = (_Float16*)alloc((size_t)HID * HID * 2);
    _Float16* Wo16 = (_Float16*)alloc((size_t)HID * HID * 2);
    _Float16* Qb   = (_Float16*)alloc(MTOK * HID * 2);
    _Float16* Kb   = (_Float16*)alloc(MTOK * HID * 2);
    _Float16* Vtb  = (_Float16*)alloc(MTOK * HID * 2);
    _Float16* CTX  = (_Float16*)alloc(MTOK * HID * 2);
    float*    PRE  = (float*)alloc(MTOK * HID * 4);
    (void)ws_size; (void)n_in; (void)in_sizes; (void)out_size;

    const int nX = (int)(MTOK * HID);
    const int nW = HID * HID;
    cvt_f32_to_f16<<<(nX + 255) / 256, 256, 0, stream>>>(hidden, X16, nX);
    cvt_f32_to_f16<<<(nW + 255) / 256, 256, 0, stream>>>(Wq, Wq16, nW);
    cvt_f32_to_f16<<<(nW + 255) / 256, 256, 0, stream>>>(Wk, Wk16, nW);
    cvt_f32_to_f16<<<(nW + 255) / 256, 256, 0, stream>>>(Wv, Wv16, nW);
    cvt_f32_to_f16<<<(nW + 255) / 256, 256, 0, stream>>>(Wo, Wo16, nW);

    // QKV projections: M=8192, N=1024; WG tile 128x256
    gemm_qkv_kernel<<<dim3(HID / 256, MTOK / 128, 3), 256, 0, stream>>>(
        X16, Wq16, Wk16, Wv16, bq, bk, bv, Qb, Kb, Vtb);

    // flash attention: one wave per (b, h, 64-row q-tile)
    attn_kernel<<<dim3(SEQ / 64, NH, 4), 32, 0, stream>>>(
        Qb, Kb, Vtb, mask, CTX);

    // output projection + bias + residual
    gemm_out_kernel<<<dim3(HID / 256, MTOK / 128, 1), 256, 0, stream>>>(
        CTX, Wo16, bo, hidden, PRE);

    // LayerNorm
    ln_kernel<<<(int)MTOK, 256, 0, stream>>>(PRE, ln_g, ln_b, (float*)d_out);
}
